// FMAF_6253472383185
// MI455X (gfx1250) — matrix operand order repaired and checked by reading, not compiled, not run on hardware
//
#include <hip/hip_runtime.h>

typedef __attribute__((ext_vector_type(8)))  float  v8f;
typedef __attribute__((ext_vector_type(16))) __bf16 v16bf;
typedef __attribute__((ext_vector_type(8)))  __bf16 v8bf;

#define BATCH 2
#define CCH   48
#define DD    32
#define HH    64
#define WW    64
#define SSP   131072        // D*H*W
#define HWP   4096          // H*W
#define NT    12582912UL    // B*C*S
#define HEADS 8
#define CP    6
#define FF    24576         // H*W*CP
#define XSTR  72            // LDS row stride (halves) for 64-deep K tiles

__device__ inline v8f vzero8() {
    v8f z;
#pragma unroll
    for (int i = 0; i < 8; i++) z[i] = 0.f;
    return z;
}
__device__ inline v16bf cat8(v8bf lo, v8bf hi) {
    return __builtin_shufflevector(lo, hi, 0,1,2,3,4,5,6,7,8,9,10,11,12,13,14,15);
}
__device__ inline v8f wmma_bf16(v16bf a, v16bf b, v8f c) {
    return __builtin_amdgcn_wmma_f32_16x16x32_bf16(false, a, false, b, (short)0, c, false, false);
}

// ---------------- LayerNorm over channels (axis=1) ----------------
__global__ void k_ln(const float* __restrict__ x, const float* __restrict__ w,
                     const float* __restrict__ bb, float* __restrict__ y) {
    long i = (long)blockIdx.x * 256 + threadIdx.x;   // 0 .. B*S-1
    int b = (int)(i >> 17);
    int s = (int)(i & (SSP - 1));
    const float* base = x + (size_t)b * CCH * SSP + s;
    float v[CCH];
    float mu = 0.f;
#pragma unroll
    for (int c = 0; c < CCH; c++) { v[c] = base[(size_t)c * SSP]; mu += v[c]; }
    mu *= (1.f / CCH);
    float var = 0.f;
#pragma unroll
    for (int c = 0; c < CCH; c++) { float d = v[c] - mu; var += d * d; }
    float inv = rsqrtf(var * (1.f / CCH) + 1e-5f);
    float* ob = y + (size_t)b * CCH * SSP + s;
#pragma unroll
    for (int c = 0; c < CCH; c++) ob[(size_t)c * SSP] = (v[c] - mu) * inv * w[c] + bb[c];
}

// ------------- circulant kernels of the ideal band mask -------------
__global__ void k_circ(float* c32o, float* c64o) {
    int n = threadIdx.x;  // 64 threads
    if (n < 32) {
        float s = 0.f;
        for (int j = 8; j < 24; j++) s += __cosf(6.28318530718f * j * n / 32.f);
        c32o[n] = s * (1.f / 32.f);
    }
    float s = 0.f;
    for (int j = 16; j < 48; j++) s += __cosf(6.28318530718f * j * n / 64.f);
    c64o[n] = s * (1.f / 64.f);
}

// ------- fused circular conv along W then H for one (b,c,d) slice -------
__global__ void k_hw(const float* __restrict__ src, float* __restrict__ dst,
                     const float* __restrict__ c64) {
    __shared__ float tA[64][65];
    __shared__ float tB[64][65];
    __shared__ float ker[64];
    int tid = threadIdx.x;
    size_t base = (size_t)blockIdx.x * HWP;
    if (tid < 64) ker[tid] = c64[tid];
    for (int q = tid; q < HWP; q += 256) tA[q >> 6][q & 63] = src[base + q];
    __builtin_prefetch(src + base + HWP + tid * 4, 0, 1);   // stream next slice
    __syncthreads();
    for (int q = tid; q < HWP; q += 256) {
        int h = q >> 6, w = q & 63;
        float acc = 0.f;
        for (int n = 0; n < 64; n++) acc += ker[n] * tA[h][(w - n) & 63];
        tB[h][w] = acc;
    }
    __syncthreads();
    for (int q = tid; q < HWP; q += 256) {
        int h = q >> 6, w = q & 63;
        float acc = 0.f;
        for (int n = 0; n < 64; n++) acc += ker[n] * tB[(h - n) & 63][w];
        dst[base + q] = acc;
    }
}

// --- circular conv along D + combine: out = alpha*orig + beta*low ---
__global__ void k_dcomb(const float* __restrict__ lowhw, const float* __restrict__ orig,
                        float* __restrict__ out, const float* __restrict__ c32,
                        float alpha, float beta) {
    __shared__ float t[32][64];
    __shared__ float ker[32];
    int tid = threadIdx.x;
    int h  = blockIdx.x & 63;
    int bc = blockIdx.x >> 6;
    size_t base = (size_t)bc * SSP + h * 64;
    if (tid < 32) ker[tid] = c32[tid];
    for (int q = tid; q < 2048; q += 256) {
        int d = q >> 6, w = q & 63;
        t[d][w] = lowhw[base + (size_t)d * HWP + w];
    }
    __syncthreads();
    for (int q = tid; q < 2048; q += 256) {
        int d = q >> 6, w = q & 63;
        float acc = 0.f;
        for (int n = 0; n < 32; n++) acc += ker[n] * t[(d - n) & 31][w];
        size_t g = base + (size_t)d * HWP + w;
        out[g] = alpha * orig[g] + beta * acc;
    }
}

// ------ sum of three spatial depthwise convs (7,11,21) per (b,c,d) ------
__global__ void k_dws(const float* __restrict__ src, float* __restrict__ dst,
                      const float* __restrict__ w7,  const float* __restrict__ b7,
                      const float* __restrict__ w11, const float* __restrict__ b11,
                      const float* __restrict__ w21, const float* __restrict__ b21) {
    __shared__ float t[84 * 84];
    __shared__ float wl[611];
    __shared__ float bsum;
    int tid = threadIdx.x;
    int bcd = blockIdx.x;             // (b*48+c)*32+d
    int c = (bcd >> 5) % CCH;
    size_t base = (size_t)bcd * HWP;
    for (int q = tid; q < 611; q += 256) {
        float v;
        if (q < 49)       v = w7 [c * 49  + q];
        else if (q < 170) v = w11[c * 121 + (q - 49)];
        else              v = w21[c * 441 + (q - 170)];
        wl[q] = v;
    }
    if (tid == 0) bsum = b7[c] + b11[c] + b21[c];
    for (int q = tid; q < 7056; q += 256) {
        int ty = q / 84, tx = q - ty * 84;
        int h = ty - 10, w = tx - 10;
        float v = 0.f;
        if (h >= 0 && h < 64 && w >= 0 && w < 64) v = src[base + h * 64 + w];
        t[q] = v;
    }
    __builtin_prefetch(src + base + HWP + tid * 4, 0, 1);
    __syncthreads();
    for (int q = tid; q < HWP; q += 256) {
        int h = q >> 6, w = q & 63;
        float acc = bsum;
        for (int dy = 0; dy < 7; dy++) {
            int yy = h + 7 + dy;
            for (int dx = 0; dx < 7; dx++) acc += wl[dy * 7 + dx] * t[yy * 84 + w + 7 + dx];
        }
        for (int dy = 0; dy < 11; dy++) {
            int yy = h + 5 + dy;
            for (int dx = 0; dx < 11; dx++) acc += wl[49 + dy * 11 + dx] * t[yy * 84 + w + 5 + dx];
        }
        for (int dy = 0; dy < 21; dy++) {
            int yy = h + dy;
            for (int dx = 0; dx < 21; dx++) acc += wl[170 + dy * 21 + dx] * t[yy * 84 + w + dx];
        }
        dst[base + q] = acc;
    }
}

// ------ accumulate three temporal depthwise convs (7,11,21) along D ------
__global__ void k_dwt(const float* __restrict__ src, float* __restrict__ dst,
                      const float* __restrict__ w7,  const float* __restrict__ b7,
                      const float* __restrict__ w11, const float* __restrict__ b11,
                      const float* __restrict__ w21, const float* __restrict__ b21) {
    __shared__ float t[52 * 64];
    __shared__ float wl[39];
    __shared__ float bsum;
    int tid = threadIdx.x;
    int h  = blockIdx.x & 63;
    int bc = blockIdx.x >> 6;
    int c = bc % CCH;
    size_t base = (size_t)bc * SSP + h * 64;
    if (tid < 39)
        wl[tid] = (tid < 7) ? w7[c * 7 + tid]
                 : ((tid < 18) ? w11[c * 11 + tid - 7] : w21[c * 21 + tid - 18]);
    if (tid == 0) bsum = b7[c] + b11[c] + b21[c];
    for (int q = tid; q < 3328; q += 256) {
        int ty = q >> 6, w = q & 63;
        int d = ty - 10;
        t[q] = (d >= 0 && d < 32) ? src[base + (size_t)d * HWP + w] : 0.f;
    }
    __syncthreads();
    for (int q = tid; q < 2048; q += 256) {
        int d = q >> 6, w = q & 63;
        float acc = bsum;
        for (int u = 0; u < 7;  u++) acc += wl[u]      * t[(d + 7 + u)  * 64 + w];
        for (int u = 0; u < 11; u++) acc += wl[7 + u]  * t[(d + 5 + u)  * 64 + w];
        for (int u = 0; u < 21; u++) acc += wl[18 + u] * t[(d + u)      * 64 + w];
        size_t g = base + (size_t)d * HWP + w;
        dst[g] += acc;
    }
}

// ---- 1x1x1 projection: Y(48 x S) = W(48x48) @ X(48 x S) + bias, via WMMA bf16 ----
// Block = 8 waves = 128 spatial positions. X tile staged to LDS (K padded to 64),
// so fragment builds are branch-free ds_load_b128 pairs and global reads coalesce.
__global__ void k_proj(const float* __restrict__ src, float* __restrict__ dst,
                       const float* __restrict__ pw, const float* __restrict__ pb) {
    __shared__ __attribute__((aligned(32))) __bf16 wl[48 * 64];
    __shared__ __attribute__((aligned(32))) __bf16 xt[128 * XSTR];
    int tid = threadIdx.x;
    int b    = blockIdx.x >> 10;
    int sblk = (blockIdx.x & 1023) << 7;
    for (int q = tid; q < 3072; q += 256) {
        int o = q >> 6, k = q & 63;
        wl[q] = (k < 48) ? (__bf16)pw[o * 48 + k] : (__bf16)0.f;
    }
    for (int q = tid; q < 48 * 128; q += 256) {          // coalesced: 128-float rows
        int c = q >> 7, j = q & 127;
        xt[j * XSTR + c] = (__bf16)src[((size_t)b * 48 + c) * SSP + sblk + j];
    }
    for (int q = tid; q < 16 * 128; q += 256) {          // zero K padding 48..63
        int k = 48 + (q >> 7), j = q & 127;
        xt[j * XSTR + k] = (__bf16)0.f;
    }
    __syncthreads();
    int lane = tid & 31;
    int hi8  = (lane & 16) ? 8 : 0;
    int j    = ((tid >> 5) << 4) + (lane & 15);          // local spatial column
    v8f acc[3];
    acc[0] = vzero8(); acc[1] = vzero8(); acc[2] = vzero8();
#pragma unroll
    for (int kk = 0; kk < 2; kk++) {
        const __bf16* xr = &xt[j * XSTR + kk * 32];
        v16bf bf = cat8(*(const v8bf*)(xr + hi8), *(const v8bf*)(xr + 16 + hi8));
#pragma unroll
        for (int m = 0; m < 3; m++) {
            const __bf16* wr = &wl[(m * 16 + (lane & 15)) * 64 + kk * 32];
            v16bf af = cat8(*(const v8bf*)(wr + hi8), *(const v8bf*)(wr + 16 + hi8));
            acc[m] = wmma_bf16(af, bf, acc[m]);
        }
    }
    int s = sblk + j;
#pragma unroll
    for (int m = 0; m < 3; m++)
#pragma unroll
        for (int r = 0; r < 8; r++) {
            int o = m * 16 + r + hi8;
            dst[((size_t)b * 48 + o) * SSP + s] = acc[m][r] + pb[o];
        }
}

// ---- pack to heads layout: PH[b,hd,d,f] (f = hw*6+cp), bf16 ----
__global__ void k_pack(const float* __restrict__ src, __bf16* __restrict__ dst) {
    __shared__ __attribute__((aligned(32))) __bf16 lb[6 * 4096];
    int tid = threadIdx.x;
    int blk = blockIdx.x;              // (b*8+hd)*32 + d
    int d  = blk & 31;
    int bh = blk >> 5;
    int b = bh >> 3, hd = bh & 7;
    for (int c = 0; c < 6; c++) {
        size_t sb = ((size_t)(b * 48 + hd * 6 + c) * 32 + d) * HWP;
        for (int q = tid; q < HWP; q += 256) lb[c * 4096 + q] = (__bf16)src[sb + q];
    }
    __syncthreads();
    size_t db = (size_t)blk * FF;
    for (int f = tid; f < FF; f += 256) {
        int hw = f / 6;
        int cp = f - hw * 6;
        dst[db + f] = lb[cp * 4096 + hw];
    }
}

// ---- reciprocal L2 norms over f per (b,hd,d) ----
__global__ void k_norm(const __bf16* __restrict__ P, float* __restrict__ invn) {
    __shared__ float red[256];
    int tid = threadIdx.x;
    size_t base = (size_t)blockIdx.x * FF;
    float s = 0.f;
    for (int f = tid; f < FF; f += 256) { float v = (float)P[base + f]; s += v * v; }
    red[tid] = s;
    __syncthreads();
    for (int st = 128; st > 0; st >>= 1) {
        if (tid < st) red[tid] += red[tid + st];
        __syncthreads();
    }
    if (tid == 0) invn[blockIdx.x] = 1.f / fmaxf(sqrtf(red[0]), 1e-12f);
}

// ---- Gram: G[bh,d,e] = sum_f P2H[d,f] * P1H[e,f], WMMA bf16, K split over 8 waves ----
__global__ void k_gram(const __bf16* __restrict__ P2, const __bf16* __restrict__ P1,
                       float* __restrict__ G) {
    __shared__ float red[8][4][32][8];   // 32 KB
    int tid = threadIdx.x, lane = tid & 31, wv = tid >> 5;
    int bh = blockIdx.x;
    const __bf16* A  = P2 + (size_t)bh * 32 * FF;
    const __bf16* Bm = P1 + (size_t)bh * 32 * FF;
    v8f acc[2][2];
    acc[0][0] = vzero8(); acc[0][1] = vzero8(); acc[1][0] = vzero8(); acc[1][1] = vzero8();
    int k0 = wv * (FF / 8);
    for (int kk = k0; kk < k0 + FF / 8; kk += 32) {
        v16bf af[2], bf[2];
#pragma unroll
        for (int t = 0; t < 2; t++) {
            int row = t * 16 + (lane & 15);
            const __bf16* pr = A + (size_t)row * FF + kk;
            af[t] = cat8(*(const v8bf*)(pr + ((lane & 16) ? 8 : 0)),
                         *(const v8bf*)(pr + 16 + ((lane & 16) ? 8 : 0)));
            int col = t * 16 + (lane & 15);
            const __bf16* pc = Bm + (size_t)col * FF + kk;
            bf[t] = cat8(*(const v8bf*)(pc + ((lane & 16) ? 8 : 0)),
                         *(const v8bf*)(pc + 16 + ((lane & 16) ? 8 : 0)));
        }
#pragma unroll
        for (int mt = 0; mt < 2; mt++)
#pragma unroll
            for (int nt = 0; nt < 2; nt++)
                acc[mt][nt] = wmma_bf16(af[mt], bf[nt], acc[mt][nt]);
    }
#pragma unroll
    for (int mt = 0; mt < 2; mt++)
#pragma unroll
        for (int nt = 0; nt < 2; nt++)
#pragma unroll
            for (int r = 0; r < 8; r++) red[wv][mt * 2 + nt][lane][r] = acc[mt][nt][r];
    __syncthreads();
    for (int q = tid; q < 1024; q += 256) {
        int t = q >> 8, ln = (q >> 3) & 31, r = q & 7;
        float s = 0.f;
        for (int w = 0; w < 8; w++) s += red[w][t][ln][r];
        int mt = t >> 1, nt = t & 1;
        int d = mt * 16 + r + ((ln & 16) ? 8 : 0);
        int e = nt * 16 + (ln & 15);
        G[(size_t)bh * 1024 + d * 32 + e] = s;
    }
}

// ---- row softmax of scaled Gram for both attention directions ----
__global__ void k_smax(const float* __restrict__ G, const float* __restrict__ invn1,
                       const float* __restrict__ invn2,
                       __bf16* __restrict__ A1, __bf16* __restrict__ A2) {
    int bh = blockIdx.x;
    int d = threadIdx.x;                 // 32 threads
    const float* g = G + (size_t)bh * 1024;
    float r[32];
    // A1[d,e] = softmax_e( G[d,e] * invn2[d] * invn1[e] )
    float qd = invn2[bh * 32 + d];
    float mx = -1e30f;
#pragma unroll
    for (int e = 0; e < 32; e++) { r[e] = g[d * 32 + e] * qd * invn1[bh * 32 + e]; mx = fmaxf(mx, r[e]); }
    float s = 0.f;
#pragma unroll
    for (int e = 0; e < 32; e++) { r[e] = __expf(r[e] - mx); s += r[e]; }
    float inv = 1.f / s;
#pragma unroll
    for (int e = 0; e < 32; e++) A1[(size_t)bh * 1024 + d * 32 + e] = (__bf16)(r[e] * inv);
    // A2[d,e] = softmax_e( G[e,d] * invn1[d] * invn2[e] )
    float qd2 = invn1[bh * 32 + d];
    mx = -1e30f;
#pragma unroll
    for (int e = 0; e < 32; e++) { r[e] = g[e * 32 + d] * qd2 * invn2[bh * 32 + e]; mx = fmaxf(mx, r[e]); }
    s = 0.f;
#pragma unroll
    for (int e = 0; e < 32; e++) { r[e] = __expf(r[e] - mx); s += r[e]; }
    inv = 1.f / s;
#pragma unroll
    for (int e = 0; e < 32; e++) A2[(size_t)bh * 1024 + d * 32 + e] = (__bf16)(r[e] * inv);
}

// ---- AV: out3 = A1 @ P1H + P2H*invn2 ; out4 = A2 @ P2H + P1H*invn1 (WMMA) ----
// Block = 8 waves = 128 f-columns of one (b,head); P1/P2 tiles staged in LDS so the
// B fragments and normalized-q epilogue avoid 24576-stride global gathers.
__global__ void k_av(const __bf16* __restrict__ P1, const __bf16* __restrict__ P2,
                     const __bf16* __restrict__ A1, const __bf16* __restrict__ A2,
                     const float* __restrict__ invn1, const float* __restrict__ invn2,
                     __bf16* __restrict__ O3, __bf16* __restrict__ O4) {
    __shared__ __attribute__((aligned(32))) __bf16 t1[32 * 128];
    __shared__ __attribute__((aligned(32))) __bf16 t2[32 * 128];
    int tid = threadIdx.x, lane = tid & 31;
    int bh   = blockIdx.x / 192;
    int fblk = (blockIdx.x - bh * 192) * 128;
    const __bf16* p1 = P1 + (size_t)bh * 32 * FF;
    const __bf16* p2 = P2 + (size_t)bh * 32 * FF;
    for (int q = tid; q < 4096; q += 256) {              // coalesced 128-half rows
        int e = q >> 7, jj = q & 127;
        size_t g = (size_t)e * FF + fblk + jj;
        t1[q] = p1[g];
        t2[q] = p2[g];
    }
    __syncthreads();
    v16bf a1f[2], a2f[2];
#pragma unroll
    for (int mt = 0; mt < 2; mt++) {
        int row = mt * 16 + (lane & 15);
        const __bf16* r1 = A1 + (size_t)bh * 1024 + row * 32;
        const __bf16* r2 = A2 + (size_t)bh * 1024 + row * 32;
        int o = (lane & 16) ? 8 : 0;
        a1f[mt] = cat8(*(const v8bf*)(r1 + o), *(const v8bf*)(r1 + 16 + o));
        a2f[mt] = cat8(*(const v8bf*)(r2 + o), *(const v8bf*)(r2 + 16 + o));
    }
    int jloc = ((tid >> 5) << 4) + (lane & 15);
    int fcol = fblk + jloc;
    v16bf b1, b2;
#pragma unroll
    for (int i = 0; i < 16; i++) {
        int e = (i & 7) + ((i & 8) ? 16 : 0) + ((lane & 16) ? 8 : 0);
        b1[i] = t1[e * 128 + jloc];
        b2[i] = t2[e * 128 + jloc];
    }
#pragma unroll
    for (int mt = 0; mt < 2; mt++) {
        v8f d3 = wmma_bf16(a1f[mt], b1, vzero8());
        v8f d4 = wmma_bf16(a2f[mt], b2, vzero8());
#pragma unroll
        for (int r = 0; r < 8; r++) {
            int d = mt * 16 + r + ((lane & 16) ? 8 : 0);
            float q3 = (float)t2[d * 128 + jloc] * invn2[bh * 32 + d];
            float q4 = (float)t1[d * 128 + jloc] * invn1[bh * 32 + d];
            O3[((size_t)bh * 32 + d) * FF + fcol] = (__bf16)(d3[r] + q3);
            O4[((size_t)bh * 32 + d) * FF + fcol] = (__bf16)(d4[r] + q4);
        }
    }
}

// ---- final: out = Wp @ (out3+out4) + 2*bias + x1' + x2'  ----
// The from-heads gather is restructured: per head, 128 spatial columns are 768
// contiguous f values, read coalesced and de-interleaved (/6) into the LDS tile.
__global__ void k_final(const __bf16* __restrict__ O3, const __bf16* __restrict__ O4,
                        const float* __restrict__ pw, const float* __restrict__ pb,
                        const float* __restrict__ R0, const float* __restrict__ R1,
                        float* __restrict__ out) {
    __shared__ __attribute__((aligned(32))) __bf16 wl[48 * 64];
    __shared__ __attribute__((aligned(32))) __bf16 xt[128 * XSTR];
    int tid = threadIdx.x;
    int b    = blockIdx.x >> 10;
    int sblk = (blockIdx.x & 1023) << 7;
    int d    = sblk >> 12;
    int hw0  = sblk & 4095;
    for (int q = tid; q < 3072; q += 256) {
        int o = q >> 6, k = q & 63;
        wl[q] = (k < 48) ? (__bf16)pw[o * 48 + k] : (__bf16)0.f;
    }
    for (int hd = 0; hd < 8; hd++) {                      // coalesced 768-half runs
        size_t base = ((size_t)(b * 8 + hd) * 32 + d) * FF + (size_t)hw0 * 6;
        for (int q = tid; q < 768; q += 256) {
            float v = (float)O3[base + q] + (float)O4[base + q];
            int jj = q / 6;
            int cp = q - jj * 6;
            xt[jj * XSTR + hd * 6 + cp] = (__bf16)v;
        }
    }
    for (int q = tid; q < 16 * 128; q += 256) {           // zero K padding 48..63
        int k = 48 + (q >> 7), jj = q & 127;
        xt[jj * XSTR + k] = (__bf16)0.f;
    }
    __syncthreads();
    int lane = tid & 31;
    int hi8  = (lane & 16) ? 8 : 0;
    int j    = ((tid >> 5) << 4) + (lane & 15);
    v8f acc[3];
    acc[0] = vzero8(); acc[1] = vzero8(); acc[2] = vzero8();
#pragma unroll
    for (int kk = 0; kk < 2; kk++) {
        const __bf16* xr = &xt[j * XSTR + kk * 32];
        v16bf bf = cat8(*(const v8bf*)(xr + hi8), *(const v8bf*)(xr + 16 + hi8));
#pragma unroll
        for (int m = 0; m < 3; m++) {
            const __bf16* wr = &wl[(m * 16 + (lane & 15)) * 64 + kk * 32];
            v16bf af = cat8(*(const v8bf*)(wr + hi8), *(const v8bf*)(wr + 16 + hi8));
            acc[m] = wmma_bf16(af, bf, acc[m]);
        }
    }
    int s = sblk + j;
#pragma unroll
    for (int m = 0; m < 3; m++)
#pragma unroll
        for (int r = 0; r < 8; r++) {
            int o = m * 16 + r + hi8;
            size_t g = ((size_t)b * 48 + o) * SSP + s;
            out[g] = acc[m][r] + 2.f * pb[o] + R0[g] + R1[g];
        }
}

extern "C" void kernel_launch(void* const* d_in, const int* in_sizes, int n_in,
                              void* d_out, int out_size, void* d_ws, size_t ws_size,
                              hipStream_t stream) {
    const float* x1   = (const float*)d_in[0];
    const float* x2   = (const float*)d_in[1];
    const float* n1w  = (const float*)d_in[2];
    const float* n1b  = (const float*)d_in[3];
    const float* n2w  = (const float*)d_in[4];
    const float* n2b  = (const float*)d_in[5];
    const float* pw   = (const float*)d_in[6];
    const float* pb   = (const float*)d_in[7];
    // dict order: spatial c11{1,2,3}, c21{1,2,3}, then temporal c12{1,2,3}, c22{1,2,3}
    const float* c111w = (const float*)d_in[8];   const float* c111b = (const float*)d_in[9];
    const float* c112w = (const float*)d_in[10];  const float* c112b = (const float*)d_in[11];
    const float* c113w = (const float*)d_in[12];  const float* c113b = (const float*)d_in[13];
    const float* c211w = (const float*)d_in[14];  const float* c211b = (const float*)d_in[15];
    const float* c212w = (const float*)d_in[16];  const float* c212b = (const float*)d_in[17];
    const float* c213w = (const float*)d_in[18];  const float* c213b = (const float*)d_in[19];
    const float* c121w = (const float*)d_in[20];  const float* c121b = (const float*)d_in[21];
    const float* c122w = (const float*)d_in[22];  const float* c122b = (const float*)d_in[23];
    const float* c123w = (const float*)d_in[24];  const float* c123b = (const float*)d_in[25];
    const float* c221w = (const float*)d_in[26];  const float* c221b = (const float*)d_in[27];
    const float* c222w = (const float*)d_in[28];  const float* c222b = (const float*)d_in[29];
    const float* c223w = (const float*)d_in[30];  const float* c223b = (const float*)d_in[31];

    float* ws = (float*)d_ws;
    float* W0 = ws;               // x1' (kept for residual)
    float* W1 = ws + NT;          // x2'
    float* W2 = ws + 2 * NT;      // scratch / later P1H+P2H (bf16)
    float* W3 = ws + 3 * NT;      // p1 / later O3H (bf16)
    float* W4 = ws + 4 * NT;      // p2 / later O4H (bf16)
    float* SM = ws + 5 * NT;
    float* c32   = SM;
    float* c64   = SM + 32;
    float* invn1 = SM + 96;
    float* invn2 = SM + 608;
    float* G     = SM + 1120;
    __bf16* A1h  = (__bf16*)(SM + 1120 + 16384);
    __bf16* A2h  = A1h + 16384;
    __bf16* P1H  = (__bf16*)W2;
    __bf16* P2H  = P1H + NT;
    __bf16* O3H  = (__bf16*)W3;
    __bf16* O4H  = (__bf16*)W4;
    float* outp  = (float*)d_out;

    // 1. LayerNorm
    k_ln<<<1024, 256, 0, stream>>>(x1, n1w, n1b, W0);
    k_ln<<<1024, 256, 0, stream>>>(x2, n2w, n2b, W1);
    // 2. circulant kernels of the FFT band mask
    k_circ<<<1, 64, 0, stream>>>(c32, c64);
    // 3. freq split + combine:  x1' = t1 + low1 ;  x2' = 2*t2 - low2  (= t2 + high2)
    k_hw<<<3072, 256, 0, stream>>>(W0, W2, c64);
    k_dcomb<<<6144, 256, 0, stream>>>(W2, W0, W0, c32, 1.f, 1.f);
    k_hw<<<3072, 256, 0, stream>>>(W1, W2, c64);
    k_dcomb<<<6144, 256, 0, stream>>>(W2, W1, W1, c32, 2.f, -1.f);
    // 4. depthwise conv sums + projections
    k_dws<<<3072, 256, 0, stream>>>(W0, W2, c111w, c111b, c112w, c112b, c113w, c113b);
    k_dwt<<<6144, 256, 0, stream>>>(W0, W2, c121w, c121b, c122w, c122b, c123w, c123b);
    k_proj<<<2048, 256, 0, stream>>>(W2, W3, pw, pb);
    k_dws<<<3072, 256, 0, stream>>>(W1, W2, c211w, c211b, c212w, c212b, c213w, c213b);
    k_dwt<<<6144, 256, 0, stream>>>(W1, W2, c221w, c221b, c222w, c222b, c223w, c223b);
    k_proj<<<2048, 256, 0, stream>>>(W2, W4, pw, pb);
    // 5. heads layout (bf16) + norms
    k_pack<<<512, 256, 0, stream>>>(W3, P1H);
    k_pack<<<512, 256, 0, stream>>>(W4, P2H);
    k_norm<<<512, 256, 0, stream>>>(P1H, invn1);
    k_norm<<<512, 256, 0, stream>>>(P2H, invn2);
    // 6. shared Gram + dual softmax + AV with normalized-q residual
    k_gram<<<16, 256, 0, stream>>>(P2H, P1H, G);
    k_smax<<<16, 32, 0, stream>>>(G, invn1, invn2, A1h, A2h);
    k_av<<<3072, 256, 0, stream>>>(P1H, P2H, A1h, A2h, invn1, invn2, O3H, O4H);
    // 7. fused final: proj(out3)+proj(out4)+x1'+x2' = Wp@(o3+o4) + 2b + residuals
    k_final<<<2048, 256, 0, stream>>>(O3H, O4H, pw, pb, W0, W1, outp);
}
